// VADNet_43069932044326
// MI455X (gfx1250) — matrix operand order repaired and checked by reading, not compiled, not run on hardware
//
#include <hip/hip_runtime.h>

typedef __attribute__((ext_vector_type(16))) _Float16 v16h;
typedef __attribute__((ext_vector_type(8)))  _Float16 v8h;
typedef __attribute__((ext_vector_type(4)))  _Float16 v4h;
typedef __attribute__((ext_vector_type(8)))  float    v8f;
typedef __attribute__((ext_vector_type(4)))  float    v4f;

#define BB 256
#define TT 3000
#define II 40
#define HH 64
#define GG 256    // 4*H gate columns
#define KC 128    // padded concat K: [x(40) | pad(24) | h(64)]
#define ALD 136   // a_lds row stride (halves)  -> 272B, bank-conflict padded
#define GLD 17    // gates_lds row stride (floats) -> 68B, bank-conflict padded

#define LOG2E 1.4426950408889634f

// branch-free sigmoid: v_exp_f32 + v_rcp_f32 (inf -> rcp -> 0, saturates cleanly)
__device__ __forceinline__ float sigf(float x) {
    return __builtin_amdgcn_rcpf(1.0f + __builtin_amdgcn_exp2f(-LOG2E * x));
}

// native v_tanh_f32 on gfx1250 if the toolchain exposes it, else exp2+rcp
__device__ __forceinline__ float tanh_fast(float x) {
#if __has_builtin(__builtin_amdgcn_tanhf)
    return __builtin_amdgcn_tanhf(x);
#else
    // tanh(x) = 1 - 2/(exp(2x)+1); exp overflow -> rcp(inf)=0 -> 1, underflow -> -1
    float e = __builtin_amdgcn_exp2f(2.0f * LOG2E * x);
    return 1.0f - 2.0f * __builtin_amdgcn_rcpf(e + 1.0f);
#endif
}

// ---------------------------------------------------------------------------
// Kernel A: one direction-tile of the LSTM scan per workgroup.
//   grid = 32 blocks: blockIdx>>4 = direction, blockIdx&15 = batch tile (M=16)
//   block = 256 threads = 8 waves; wave w owns gate columns [w*32, w*32+32)
// ---------------------------------------------------------------------------
__global__ __launch_bounds__(256)
void lstm_dir_kernel(const float* __restrict__ x,
                     const float* __restrict__ Wih_f, const float* __restrict__ Whh_f,
                     const float* __restrict__ bih_f, const float* __restrict__ bhh_f,
                     const float* __restrict__ Wih_b, const float* __restrict__ Whh_b,
                     const float* __restrict__ bih_b, const float* __restrict__ bhh_b,
                     _Float16* __restrict__ hbuf)
{
    __shared__ __align__(16) _Float16 a_lds[16 * ALD];     // A = [x_t | h_t] f16
    __shared__ __align__(16) float    gates_lds[GG * GLD]; // gates, [N][M] layout

    const int tid  = threadIdx.x;
    const int lane = tid & 31;
    const int wv   = tid >> 5;            // wave 0..7
    const int dir  = blockIdx.x >> 4;     // 0 = fwd, 1 = bwd
    const int b0   = (blockIdx.x & 15) * 16;

    const float* Wih = dir ? Wih_b : Wih_f;
    const float* Whh = dir ? Whh_b : Whh_f;
    const float* bih = dir ? bih_b : bih_f;
    const float* bhh = dir ? bhh_b : bhh_f;

    // zero A staging: covers K-pad (40..63) and h0 == 0
    for (int i = tid; i < 16 * ALD; i += 256) a_lds[i] = (_Float16)0.0f;

    // ---- preload concat-weight B-tiles into VGPRs (exact WMMA B layout) ----
    // B-tile (32x16 f16): lane%16 = N column, elements e -> K = kt*32 + (e/8)*16 + (lane/16)*8 + e%8
    const int n0   = wv * 32;
    const int nlo  = lane & 15;
    const int khal = lane >> 4;
    v16h Bt[2][4];
    float biasv[2];
#pragma unroll
    for (int nt = 0; nt < 2; ++nt) {
        const int N = n0 + nt * 16 + nlo;
        biasv[nt] = bih[N] + bhh[N];
#pragma unroll
        for (int kt = 0; kt < 4; ++kt) {
            v16h bt;
#pragma unroll
            for (int e = 0; e < 16; ++e) {
                const int K = kt * 32 + (e >> 3) * 16 + khal * 8 + (e & 7);
                float v = 0.0f;
                if (K < II)       v = Wih[N * II + K];          // input part
                else if (K >= 64) v = Whh[N * HH + (K - 64)];   // hidden part
                bt[e] = (_Float16)v;
            }
            Bt[nt][kt] = bt;
        }
    }

    // ---- per-thread cell-update assignment: row m_u, 4 hidden cols j0.. ----
    const int m_u = tid >> 4;
    const int j0  = (tid & 15) * 4;
    float c_st[4] = {0.f, 0.f, 0.f, 0.f};

    // x loader assignment: 160 threads x 4 floats = 16 rows x 40 cols
    const int m_x = tid / 10;
    const int ix  = (tid % 10) * 4;

    // prologue: stage x for the first step
    if (tid < 160) {
        const int t0 = dir ? (TT - 1) : 0;
        const float* xp = x + ((size_t)(b0 + m_x) * TT + t0) * II + ix;
        v4f xv = *(const v4f*)xp;
        v4h xh = { (_Float16)xv[0], (_Float16)xv[1], (_Float16)xv[2], (_Float16)xv[3] };
        *(v4h*)&a_lds[m_x * ALD + ix] = xh;
    }

    const int mA  = lane & 15;    // A row for this lane
    const int hi8 = khal * 8;     // A K-chunk select (doc 16-bit A layout)

    for (int s = 0; s < TT; ++s) {
        const int t = dir ? (TT - 1 - s) : s;
        __syncthreads();   // a_lds holds [x_t | h_t]

        // ---- gates = bias + [x_t|h_t] * Wc^T  (8 WMMA per wave) ----
        v16h a[4];
#pragma unroll
        for (int kt = 0; kt < 4; ++kt) {
            const int base = mA * ALD + kt * 32 + hi8;
            v8h lo = *(const v8h*)&a_lds[base];        // K = kt*32 + khal*8 .. +7
            v8h hi = *(const v8h*)&a_lds[base + 16];   // K = +16
            v16h av;
#pragma unroll
            for (int e = 0; e < 8; ++e) { av[e] = lo[e]; av[e + 8] = hi[e]; }
            a[kt] = av;
        }
        // keep all 8 ds_load_b128 issued before the WMMA stream (one dscnt wait)
        __builtin_amdgcn_sched_barrier(0);
#pragma unroll
        for (int nt = 0; nt < 2; ++nt) {
            const float bv = biasv[nt];
            v8f acc = { bv, bv, bv, bv, bv, bv, bv, bv };   // fold biases into C
#pragma unroll
            for (int kt = 0; kt < 4; ++kt)
                acc = __builtin_amdgcn_wmma_f32_16x16x32_f16(
                        false, a[kt], false, Bt[nt][kt], (short)0, acc, false, false);
            // D layout: VGPR r -> M = 8*khal + r, N = nlo  => [N][M] LDS store
            const int ga = (n0 + nt * 16 + nlo) * GLD + khal * 8;
            v4f lo4 = { acc[0], acc[1], acc[2], acc[3] };
            v4f hi4 = { acc[4], acc[5], acc[6], acc[7] };
            *(v4f*)&gates_lds[ga]     = lo4;
            *(v4f*)&gates_lds[ga + 4] = hi4;
        }
        __syncthreads();   // gates visible

        // ---- LSTM cell update: c in fp32 registers, h -> f16 ----
        v4h hh;
#pragma unroll
        for (int dj = 0; dj < 4; ++dj) {
            const int j = j0 + dj;
            const float gi = gates_lds[(0   + j) * GLD + m_u];
            const float gf = gates_lds[(64  + j) * GLD + m_u];
            const float gg = gates_lds[(128 + j) * GLD + m_u];
            const float go = gates_lds[(192 + j) * GLD + m_u];
            float cc = sigf(gf) * c_st[dj] + sigf(gi) * tanh_fast(gg);
            c_st[dj] = cc;
            hh[dj] = (_Float16)(sigf(go) * tanh_fast(cc));
        }
        *(v4h*)&a_lds[m_u * ALD + 64 + j0] = hh;  // feed next step
        *(v4h*)(hbuf + ((size_t)(b0 + m_u) * TT + t) * KC + dir * HH + j0) = hh;

        // ---- stage x for the next step (+ prefetch t+2 into caches) ----
        if (s + 1 < TT && tid < 160) {
            const int tn = dir ? (TT - 2 - s) : (s + 1);
            const float* xp = x + ((size_t)(b0 + m_x) * TT + tn) * II + ix;
            v4f xv = *(const v4f*)xp;
            v4h xh = { (_Float16)xv[0], (_Float16)xv[1], (_Float16)xv[2], (_Float16)xv[3] };
            *(v4h*)&a_lds[m_x * ALD + ix] = xh;
            if (s + 2 < TT) {
                const int t2 = dir ? (TT - 3 - s) : (s + 2);
                __builtin_prefetch(x + ((size_t)(b0 + m_x) * TT + t2) * II + ix, 0, 0);
            }
        }
    }
}

// ---------------------------------------------------------------------------
// Kernel B: MLP head.  out = sigmoid(W2 . relu(W1 h + b1) + b2)
//   4 waves / block, each wave processes a 16-row tile per iteration:
//   A = h[16,128] f16, 16 WMMA -> z[16,64], relu*W2 + xor-shuffle reduce.
// ---------------------------------------------------------------------------
__global__ __launch_bounds__(128)
void head_kernel(const _Float16* __restrict__ hbuf,
                 const float* __restrict__ W1, const float* __restrict__ b1,
                 const float* __restrict__ W2, const float* __restrict__ b2,
                 float* __restrict__ out)
{
    const int tid  = threadIdx.x;
    const int lane = tid & 31;
    const int wv   = tid >> 5;
    const int nlo  = lane & 15;
    const int khal = lane >> 4;
    const int hi8  = khal * 8;

    // W1 resident in VGPRs in WMMA B layout: 16 tiles of [32,16]
    v16h Wt[4][4];
    float bias1[4], w2v[4];
#pragma unroll
    for (int nt = 0; nt < 4; ++nt) {
        const int N = nt * 16 + nlo;
        bias1[nt] = b1[N];
        w2v[nt]   = W2[N];
#pragma unroll
        for (int kt = 0; kt < 4; ++kt) {
            v16h w;
#pragma unroll
            for (int e = 0; e < 16; ++e) {
                const int K = kt * 32 + (e >> 3) * 16 + khal * 8 + (e & 7);
                w[e] = (_Float16)W1[N * KC + K];
            }
            Wt[nt][kt] = w;
        }
    }
    const float b2v = b2[0];

    const int ntiles = (BB * TT) / 16;   // 48000 row tiles
    for (int tile = blockIdx.x * 4 + wv; tile < ntiles; tile += gridDim.x * 4) {
        const size_t row0 = (size_t)tile * 16;
        const _Float16* ar = hbuf + (row0 + (size_t)nlo) * KC;

        v16h a[4];
#pragma unroll
        for (int kt = 0; kt < 4; ++kt) {
            v8h lo = *(const v8h*)&ar[kt * 32 + hi8];
            v8h hi = *(const v8h*)&ar[kt * 32 + 16 + hi8];
            v16h av;
#pragma unroll
            for (int e = 0; e < 8; ++e) { av[e] = lo[e]; av[e + 8] = hi[e]; }
            a[kt] = av;
        }
        // issue all 8 global_load_b128 before the WMMA stream
        __builtin_amdgcn_sched_barrier(0);

        float part[8] = {0.f,0.f,0.f,0.f,0.f,0.f,0.f,0.f};
#pragma unroll
        for (int nt = 0; nt < 4; ++nt) {
            const float bv = bias1[nt];
            v8f acc = { bv, bv, bv, bv, bv, bv, bv, bv };
#pragma unroll
            for (int kt = 0; kt < 4; ++kt)
                acc = __builtin_amdgcn_wmma_f32_16x16x32_f16(
                        false, a[kt], false, Wt[nt][kt], (short)0, acc, false, false);
#pragma unroll
            for (int r = 0; r < 8; ++r) {
                float z = acc[r];
                z = z > 0.f ? z : 0.f;          // relu
                part[r] += z * w2v[nt];         // * W2[N]
            }
        }
        // reduce over the 16 N-lanes of each half (rows M split by khal)
#pragma unroll
        for (int r = 0; r < 8; ++r) {
            part[r] += __shfl_xor(part[r], 1, 32);
            part[r] += __shfl_xor(part[r], 2, 32);
            part[r] += __shfl_xor(part[r], 4, 32);
            part[r] += __shfl_xor(part[r], 8, 32);
        }
        if (nlo == 0) {   // lane 0 -> rows 0..7, lane 16 -> rows 8..15
            float o[8];
#pragma unroll
            for (int r = 0; r < 8; ++r) o[r] = sigf(part[r] + b2v);
            float* op = out + row0 + (size_t)khal * 8;
            v4f o0 = { o[0], o[1], o[2], o[3] };
            v4f o1 = { o[4], o[5], o[6], o[7] };
            *(v4f*)op       = o0;
            *(v4f*)(op + 4) = o1;
        }
    }
}

// ---------------------------------------------------------------------------
extern "C" void kernel_launch(void* const* d_in, const int* in_sizes, int n_in,
                              void* d_out, int out_size, void* d_ws, size_t ws_size,
                              hipStream_t stream) {
    const float* x      = (const float*)d_in[0];
    const float* Wih_f  = (const float*)d_in[1];
    const float* Whh_f  = (const float*)d_in[2];
    const float* bih_f  = (const float*)d_in[3];
    const float* bhh_f  = (const float*)d_in[4];
    const float* Wih_b  = (const float*)d_in[5];
    const float* Whh_b  = (const float*)d_in[6];
    const float* bih_b  = (const float*)d_in[7];
    const float* bhh_b  = (const float*)d_in[8];
    const float* W1     = (const float*)d_in[9];
    const float* b1     = (const float*)d_in[10];
    const float* W2     = (const float*)d_in[11];
    const float* b2     = (const float*)d_in[12];

    _Float16* hbuf = (_Float16*)d_ws;   // B*T*128 f16 = ~197 MB

    lstm_dir_kernel<<<dim3(32), dim3(256), 0, stream>>>(
        x, Wih_f, Whh_f, bih_f, bhh_f, Wih_b, Whh_b, bih_b, bhh_b, hbuf);

    head_kernel<<<dim3(2048), dim3(128), 0, stream>>>(
        hbuf, W1, b1, W2, b2, (float*)d_out);
}
